// CharLSTM_75067438399632
// MI455X (gfx1250) — matrix operand order repaired and checked by reading, not compiled, not run on hardware
//
#include <hip/hip_runtime.h>

// CharLSTM for MI455X (gfx1250):
//  - input-gate halves hoisted into a bf16 WMMA GEMM (v_wmma_f32_16x16x32_bf16)
//  - tiles staged with GLOBAL_LOAD_ASYNC_TO_LDS_B128 (ASYNCcnt) when available,
//    128-bit vector loads otherwise
//  - recurrent matvecs stream L2-resident bf16 weights (32 MB < 192 MB L2) with
//    b128 loads in a persistent 64-block kernel, one device-wide barrier/step.

#define T_SEQ 4096
#define EMB_N 512
#define HID_N 2048
#define DIM_N 2560
#define NCH   128
#define NB_REC 64

typedef __attribute__((ext_vector_type(16))) __bf16 v16bf;
typedef __attribute__((ext_vector_type(8)))  float  v8f;
typedef __attribute__((ext_vector_type(4)))  int    i32x4;

#if defined(__has_builtin)
#if __has_builtin(__builtin_amdgcn_global_load_async_to_lds_b128) && \
    __has_builtin(__builtin_amdgcn_s_wait_asynccnt)
#define USE_ASYNC_LDS 1
#endif
#endif
#ifndef USE_ASYNC_LDS
#define USE_ASYNC_LDS 0
#endif

#define LDS_AS __attribute__((address_space(3)))
#define GLB_AS __attribute__((address_space(1)))
// builtin signature (from clang diagnostic): void(int4 AS1*, int4 AS3*, imm, imm)
#define ASYNC_B128(gsrc, ldst) \
  __builtin_amdgcn_global_load_async_to_lds_b128((GLB_AS i32x4*)(gsrc), \
                                                 (LDS_AS i32x4*)(ldst), 0, 0)

__device__ __forceinline__ unsigned short f2bf(float f) {
  unsigned x = __float_as_uint(f);
  unsigned r = (x + 0x7FFFu + ((x >> 16) & 1u)) >> 16;  // round-to-nearest-even
  return (unsigned short)r;
}
__device__ __forceinline__ float bf2f(unsigned u16) {
  return __uint_as_float(u16 << 16);
}
__device__ __forceinline__ float sigm(float x) { return 1.0f / (1.0f + expf(-x)); }
__device__ __forceinline__ void fma_pair(float& acc, unsigned p, float he, float ho) {
  acc += bf2f(p & 0xffffu) * he + bf2f(p >> 16) * ho;
}

// ---------------------------------------------------------------- init
__global__ void lstm_init(float* hbuf, float* cbuf, unsigned long long* counter) {
  if (blockIdx.x == 0 && threadIdx.x == 0) *counter = 0ull;
  for (int i = blockIdx.x * blockDim.x + threadIdx.x; i < 3 * HID_N;
       i += gridDim.x * blockDim.x) {
    if (i < 2 * HID_N) hbuf[i] = 0.0f;
    else               cbuf[i - 2 * HID_N] = 0.0f;
  }
}

// ---------------------------------------------------------------- X = emb[seq] (bf16)
__global__ void lstm_build_x(const int* __restrict__ seq,
                             const float* __restrict__ emb,
                             unsigned short* __restrict__ X) {
  int i = blockIdx.x * blockDim.x + threadIdx.x;
  if (i >= T_SEQ * EMB_N) return;
  int t = i >> 9;          // /EMB_N
  int e = i & (EMB_N - 1);
  X[i] = f2bf(emb[(size_t)seq[t] * EMB_N + e]);
}

// ---------------------------------------------------------------- split+convert weights
__global__ void lstm_convert_w(const float* __restrict__ Wf, const float* __restrict__ Wi,
                               const float* __restrict__ Wo, const float* __restrict__ Wc,
                               unsigned short* __restrict__ Wx,
                               unsigned short* __restrict__ Wh) {
  size_t i = (size_t)blockIdx.x * blockDim.x + threadIdx.x;
  const size_t per_gate = (size_t)HID_N * DIM_N;
  if (i >= 4 * per_gate) return;
  int g = (int)(i / per_gate);
  size_t rem = i - (size_t)g * per_gate;
  int h = (int)(rem / DIM_N);
  int d = (int)(rem % DIM_N);
  const float* W = (g == 0) ? Wf : (g == 1) ? Wi : (g == 2) ? Wo : Wc;
  unsigned short v = f2bf(W[(size_t)h * DIM_N + d]);
  if (d < EMB_N) Wx[((size_t)g * HID_N + h) * EMB_N + d]           = v;
  else           Wh[((size_t)g * HID_N + h) * HID_N + (d - EMB_N)] = v;
}

// ---------------------------------------------------------------- Gx = X @ Wx^T + b  (WMMA)
// grid: (T/16, H/128, 4 gates), 256 threads = 8 waves; wave w owns h-tile hbase+16w.
__global__ __launch_bounds__(256) void lstm_input_gemm(
    const __bf16* __restrict__ X, const __bf16* __restrict__ Wx,
    const float* __restrict__ bfv, const float* __restrict__ biv,
    const float* __restrict__ bov, const float* __restrict__ bcv,
    float* __restrict__ Gx) {
  __shared__ __align__(16) __bf16 Xs[16][32];          // 1 KB
  __shared__ __align__(16) __bf16 Ws[8][16][32];       // 8 KB
  const int g     = blockIdx.z;
  const int t0    = blockIdx.x * 16;
  const int hbase = blockIdx.y * 128;
  const int wave  = threadIdx.x >> 5;
  const int lane  = threadIdx.x & 31;
  const int half  = lane >> 4;   // 0: lanes 0-15, 1: lanes 16-31
  const int mn    = lane & 15;   // row (A) / col (B/D) index
  const int h0    = hbase + wave * 16;
  const float* bias = (g == 0) ? bfv : (g == 1) ? biv : (g == 2) ? bov : bcv;

  v8f acc = {};
  for (int k0 = 0; k0 < EMB_N; k0 += 32) {
#if USE_ASYNC_LDS
    {
      const int idx = threadIdx.x;
      if (idx < 64) {                         // X tile: 64 chunks x 16 B
        int m = idx >> 2, k8 = (idx & 3) * 8;
        ASYNC_B128(&X[(size_t)(t0 + m) * EMB_N + k0 + k8], &Xs[m][k8]);
      }
      for (int c = idx; c < 512; c += 256) {  // W tiles: 512 chunks x 16 B
        int w = c >> 6, n = (c >> 2) & 15, k8 = (c & 3) * 8;
        ASYNC_B128(&Wx[((size_t)g * HID_N + hbase + w * 16 + n) * EMB_N + k0 + k8],
                   &Ws[w][n][k8]);
      }
      __builtin_amdgcn_s_wait_asynccnt(0);
    }
#else
    {
      const int idx = threadIdx.x;            // X tile: 256 dword chunks
      int m = idx >> 4, k2 = idx & 15;
      ((unsigned*)&Xs[m][0])[k2] =
          ((const unsigned*)&X[(size_t)(t0 + m) * EMB_N + k0])[k2];
      for (int c = idx; c < 512; c += 256) {  // W tiles: 512 x uint4
        int w = c >> 6, n = (c >> 2) & 15, k8 = (c & 3) * 8;
        *(uint4*)&Ws[w][n][k8] =
            *(const uint4*)&Wx[((size_t)g * HID_N + hbase + w * 16 + n) * EMB_N +
                               k0 + k8];
      }
    }
#endif
    __syncthreads();
    // Fragments per CDNA5 16-bit layouts (CDNA5 ISA):
    //  A 16x32: lane<16 -> K {0..7,16..23}, lane>=16 -> K {8..15,24..31}
    //  B 32x16: lane<16 -> K {0..7,16..23}, lane>=16 -> K {8..15,24..31} (col = lane&15)
    v16bf a, b;
#pragma unroll
    for (int v = 0; v < 8; ++v) {
      int ka = (v >> 2) * 16 + half * 8 + (v & 3) * 2;
      a[2 * v]     = Xs[mn][ka];
      a[2 * v + 1] = Xs[mn][ka + 1];
      int kb = (v >> 2) * 16 + half * 8 + (v & 3) * 2;
      b[2 * v]     = Ws[wave][mn][kb];
      b[2 * v + 1] = Ws[wave][mn][kb + 1];
    }
    acc = __builtin_amdgcn_wmma_f32_16x16x32_bf16(false, a, false, b,
                                                  (short)0, acc, false, false);
    __syncthreads();
  }
  // D layout: VGPR v holds row m = v + 8*half, col = lane&15
#pragma unroll
  for (int v = 0; v < 8; ++v) {
    int m = v + half * 8;
    Gx[((size_t)g * T_SEQ + (t0 + m)) * HID_N + h0 + mn] = acc[v] + bias[h0 + mn];
  }
}

// ---------------------------------------------------------------- device-wide barrier
__device__ __forceinline__ void grid_barrier(unsigned long long* counter,
                                             unsigned nblocks,
                                             unsigned long long& epoch) {
  __syncthreads();
  epoch += 1;
  if (threadIdx.x == 0) {
    __threadfence();
    atomicAdd(counter, 1ull);
    const unsigned long long target = epoch * (unsigned long long)nblocks;
    while (atomicAdd(counter, 0ull) < target) __builtin_amdgcn_s_sleep(1);
  }
  __syncthreads();
  __threadfence();
}

// ---------------------------------------------------------------- persistent recurrence
// 64 blocks x 256 threads. Block b owns h-rows {b*32 + 8j + wave}; one wave per row
// computes all 4 gate dots (bf16 weights, b128 loads), then y rows 2b,2b+1.
__global__ __launch_bounds__(256) void lstm_recurrent(
    const uint4* __restrict__ Wh,          // bf16: 4 * H * (H/8) uint4
    const float* __restrict__ Gx, const float* __restrict__ Wy,
    const float* __restrict__ by,
    float* hbuf, float* cbuf, unsigned long long* counter,
    float* __restrict__ out) {
  __shared__ __align__(16) float hs[HID_N];
  const int block = blockIdx.x;
  const int tid   = threadIdx.x;
  const int wave  = tid >> 5;
  const int lane  = tid & 31;
  const size_t HH8 = (size_t)HID_N * (HID_N / 8);  // uint4 per gate matrix
  const int ROWQ = HID_N / 8;                      // uint4 per row
  unsigned long long epoch = 0;

  for (int t = 0; t < T_SEQ; ++t) {
    const float* hprev = hbuf + (size_t)(t & 1) * HID_N;
    float*       hnext = hbuf + (size_t)((t + 1) & 1) * HID_N;
    for (int i = tid; i < HID_N / 4; i += 256)
      ((float4*)hs)[i] = ((const float4*)hprev)[i];
    __syncthreads();

    for (int j = 0; j < 4; ++j) {
      const int r = block * 32 + j * 8 + wave;
      const uint4* w0 = Wh + 0 * HH8 + (size_t)r * ROWQ;
      const uint4* w1 = Wh + 1 * HH8 + (size_t)r * ROWQ;
      const uint4* w2 = Wh + 2 * HH8 + (size_t)r * ROWQ;
      const uint4* w3 = Wh + 3 * HH8 + (size_t)r * ROWQ;
      __builtin_prefetch(w0 + lane, 0, 0);
      __builtin_prefetch(w1 + lane, 0, 0);
      __builtin_prefetch(w2 + lane, 0, 0);
      __builtin_prefetch(w3 + lane, 0, 0);
      float a0 = 0.f, a1 = 0.f, a2 = 0.f, a3 = 0.f;
      for (int kq = lane; kq < ROWQ; kq += 32) {   // 8 bf16 per gate per iter
        const float* hp = &hs[kq * 8];
        const float e0 = hp[0], e1 = hp[1], e2 = hp[2], e3 = hp[3];
        const float e4 = hp[4], e5 = hp[5], e6 = hp[6], e7 = hp[7];
        uint4 q;
        q = w0[kq];
        fma_pair(a0, q.x, e0, e1); fma_pair(a0, q.y, e2, e3);
        fma_pair(a0, q.z, e4, e5); fma_pair(a0, q.w, e6, e7);
        q = w1[kq];
        fma_pair(a1, q.x, e0, e1); fma_pair(a1, q.y, e2, e3);
        fma_pair(a1, q.z, e4, e5); fma_pair(a1, q.w, e6, e7);
        q = w2[kq];
        fma_pair(a2, q.x, e0, e1); fma_pair(a2, q.y, e2, e3);
        fma_pair(a2, q.z, e4, e5); fma_pair(a2, q.w, e6, e7);
        q = w3[kq];
        fma_pair(a3, q.x, e0, e1); fma_pair(a3, q.y, e2, e3);
        fma_pair(a3, q.z, e4, e5); fma_pair(a3, q.w, e6, e7);
      }
#pragma unroll
      for (int off = 16; off > 0; off >>= 1) {
        a0 += __shfl_down(a0, off);
        a1 += __shfl_down(a1, off);
        a2 += __shfl_down(a2, off);
        a3 += __shfl_down(a3, off);
      }
      if (lane == 0) {
        const size_t gb = (size_t)t * HID_N + r;
        const size_t gstride = (size_t)T_SEQ * HID_N;
        float gf = a0 + Gx[0 * gstride + gb];
        float gi = a1 + Gx[1 * gstride + gb];
        float go = a2 + Gx[2 * gstride + gb];
        float gc = a3 + Gx[3 * gstride + gb];
        float f  = sigm(gf), ii = sigm(gi), o = sigm(go), cc = tanhf(gc);
        float cn = f * cbuf[r] + ii * cc;
        cbuf[r]  = cn;
        hnext[r] = o * tanhf(cn);
      }
    }

    grid_barrier(counter, NB_REC, epoch);   // h_t fully written & visible

    if (wave < 2) {                         // y_t = Wy h_t + by (2 rows per block)
      const int yr = block * 2 + wave;
      const float4* wr = (const float4*)(Wy + (size_t)yr * HID_N);
      const float4* hn = (const float4*)hnext;
      float acc = 0.f;
      for (int k = lane; k < HID_N / 4; k += 32) {
        float4 wv = wr[k], hv = hn[k];
        acc += wv.x * hv.x + wv.y * hv.y + wv.z * hv.z + wv.w * hv.w;
      }
#pragma unroll
      for (int off = 16; off > 0; off >>= 1) acc += __shfl_down(acc, off);
      if (lane == 0) out[(size_t)t * NCH + yr] = acc + by[yr];
    }
    __syncthreads();
  }

  if (lane == 0) {                          // h_last, c_last
    for (int j = 0; j < 4; ++j) {
      int r = block * 32 + j * 8 + wave;
      out[(size_t)T_SEQ * NCH + r]         = hbuf[(size_t)(T_SEQ & 1) * HID_N + r];
      out[(size_t)T_SEQ * NCH + HID_N + r] = cbuf[r];
    }
  }
}

// ---------------------------------------------------------------- launch
extern "C" void kernel_launch(void* const* d_in, const int* in_sizes, int n_in,
                              void* d_out, int out_size, void* d_ws, size_t ws_size,
                              hipStream_t stream) {
  (void)in_sizes; (void)n_in; (void)out_size; (void)ws_size;
  const int*   seq = (const int*)d_in[0];
  const float* emb = (const float*)d_in[1];
  const float* Wf  = (const float*)d_in[2];
  const float* bf_ = (const float*)d_in[3];
  const float* Wi  = (const float*)d_in[4];
  const float* bi_ = (const float*)d_in[5];
  const float* Wo  = (const float*)d_in[6];
  const float* bo_ = (const float*)d_in[7];
  const float* Wc  = (const float*)d_in[8];
  const float* bc_ = (const float*)d_in[9];
  const float* Wy  = (const float*)d_in[10];
  const float* by  = (const float*)d_in[11];
  float* out = (float*)d_out;

  char* ws = (char*)d_ws;
  size_t off = 0;
  auto alloc = [&](size_t bytes) {
    size_t o = off;
    off += (bytes + 255) & ~(size_t)255;
    return o;
  };
  unsigned long long* counter = (unsigned long long*)(ws + alloc(8));
  float*          hbuf = (float*)(ws + alloc((size_t)2 * HID_N * 4));
  float*          cbuf = (float*)(ws + alloc((size_t)HID_N * 4));
  unsigned short* X    = (unsigned short*)(ws + alloc((size_t)T_SEQ * EMB_N * 2));
  unsigned short* Wx   = (unsigned short*)(ws + alloc(4ull * HID_N * EMB_N * 2));
  unsigned short* Wh   = (unsigned short*)(ws + alloc(4ull * HID_N * HID_N * 2));
  float*          Gx   = (float*)(ws + alloc(4ull * T_SEQ * HID_N * 4));
  // total workspace ~172 MB

  lstm_init<<<16, 256, 0, stream>>>(hbuf, cbuf, counter);
  lstm_build_x<<<(T_SEQ * EMB_N) / 256, 256, 0, stream>>>(seq, emb, X);
  lstm_convert_w<<<(4 * HID_N * DIM_N) / 256, 256, 0, stream>>>(Wf, Wi, Wo, Wc, Wx, Wh);
  lstm_input_gemm<<<dim3(T_SEQ / 16, HID_N / 128, 4), 256, 0, stream>>>(
      (const __bf16*)X, (const __bf16*)Wx, bf_, bi_, bo_, bc_, Gx);
  lstm_recurrent<<<NB_REC, 256, 0, stream>>>(
      (const uint4*)Wh, Gx, Wy, by, hbuf, cbuf, counter, out);
}
